// MyModelBlock_89335319756883
// MI455X (gfx1250) — compile-verified, matrix operand order checked by reading, not run on hardware
//
#include <hip/hip_runtime.h>
#include <hip/hip_bf16.h>

#define NNODES 50000
#define NEDGES 800000
#define HD     64

typedef float v2f  __attribute__((ext_vector_type(2)));
typedef float v8f  __attribute__((ext_vector_type(8)));
typedef __bf16 v16bf __attribute__((ext_vector_type(16)));

#if defined(__has_builtin)
#if __has_builtin(__builtin_amdgcn_wmma_f32_16x16x4_f32)
#define HAVE_WMMA_F32X4 1
#endif
#endif

// ---------- helpers ----------
__device__ __forceinline__ float wave_sum(float v) {
#pragma unroll
    for (int off = 16; off > 0; off >>= 1)
        v += __shfl_xor(v, off, 32);
    return v;
}

// monotone float<->uint encoding for atomicMax on floats (incl. negatives)
__device__ __forceinline__ unsigned enc_f(float f) {
    unsigned u = __float_as_uint(f);
    return (u & 0x80000000u) ? ~u : (u | 0x80000000u);
}
__device__ __forceinline__ float dec_f(unsigned u) {
    return (u & 0x80000000u) ? __uint_as_float(u & 0x7FFFFFFFu)
                             : __uint_as_float(~u);
}
#define ENC_NEG_INF 0x007FFFFFu  // enc_f(-inf)

// ---------- v = tanh(pot @ Wpot + b), pot:(N,16) Wpot:(16,64) ----------
__global__ void pot_kernel(const float* __restrict__ pot,
                           const float* __restrict__ Wpot,
                           const float* __restrict__ bpot,
                           float* __restrict__ v) {
    int idx = blockIdx.x * 256 + threadIdx.x;     // N*64 threads
    int i = idx >> 6, c = idx & 63;
    float s = bpot[c];
    const float* pr = pot + i * 16;
#pragma unroll
    for (int k = 0; k < 16; ++k)
        s = fmaf(pr[k], Wpot[k * HD + c], s);
    v[idx] = tanhf(s);
}

// ---------- z = h@W, zi = h@U via WMMA. 1 block = 16 rows, 8 waves ----------
__global__ void __launch_bounds__(256)
gemm_kernel(const float* __restrict__ h,
            const float* __restrict__ W,
            const float* __restrict__ U,
            float* __restrict__ z,
            float* __restrict__ zi) {
    __shared__ float lds[16 * HD];  // 16-row tile of h, row-major
    // rows rb*16..rb*16+15 are contiguous 1024 floats
    {
        const float4* s4 = reinterpret_cast<const float4*>(h + (size_t)blockIdx.x * (16 * HD));
        reinterpret_cast<float4*>(lds)[threadIdx.x] = s4[threadIdx.x];
    }
    __syncthreads();

    const int wave = threadIdx.x >> 5;       // 0..7
    const int lane = threadIdx.x & 31;
    const float* Wm  = (wave < 4) ? W : U;
    float*       out = (wave < 4) ? z : zi;
    const int ct = wave & 3;                 // column tile (16 cols each)
    const int mn = lane & 15;                // row (A) / col (B) within tile

    v8f c = {};

#if defined(HAVE_WMMA_F32X4)
    // F32 16x16x4: A lane<16 holds K=0,1 ; lane>=16 holds K=2,3 (per VGPR pair)
    const int kb = (lane >> 4) << 1;         // 0 or 2
#pragma unroll
    for (int k0 = 0; k0 < HD; k0 += 4) {
        v2f a, b;
        a.x = lds[mn * HD + k0 + kb];
        a.y = lds[mn * HD + k0 + kb + 1];
        b.x = Wm[(k0 + kb)     * HD + ct * 16 + mn];
        b.y = Wm[(k0 + kb + 1) * HD + ct * 16 + mn];
        c = __builtin_amdgcn_wmma_f32_16x16x4_f32(false, a, false, b,
                                                  (short)0, c, false, false);
    }
#else
    // Fallback: bf16 16x16x32 (probe-confirmed builtin), 2 K-steps
    const int base0 = (lane >> 4) << 3;      // 0 or 8
#pragma unroll
    for (int k0 = 0; k0 < HD; k0 += 32) {
        v16bf a, b;
#pragma unroll
        for (int j = 0; j < 16; ++j) {
            int kk = k0 + base0 + j + ((j >= 8) ? 8 : 0);
            a[j] = (__bf16)lds[mn * HD + kk];
            b[j] = (__bf16)Wm[kk * HD + ct * 16 + mn];
        }
        c = __builtin_amdgcn_wmma_f32_16x16x32_bf16(false, a, false, b,
                                                    (short)0, c, false, false);
    }
#endif

    // C/D layout: VGPR r -> M=r (lanes 0-15), M=r+8 (lanes 16-31); N = lane&15
    const int rowhalf = (lane >> 4) << 3;
    const size_t rbase = (size_t)blockIdx.x * 16;
#pragma unroll
    for (int r = 0; r < 8; ++r) {
        size_t row = rbase + r + rowhalf;
        out[row * HD + ct * 16 + mn] = c[r];
    }
}

// ---------- per-node: s_src, s_dst + zero acc/denom, init segmax ----------
__global__ void nodeprep_kernel(const float* __restrict__ z,
                                const float* __restrict__ a_src,
                                const float* __restrict__ a_dst,
                                float* __restrict__ s_src,
                                float* __restrict__ s_dst,
                                float* __restrict__ acc,
                                float* __restrict__ denom,
                                unsigned* __restrict__ segmax) {
    int i = blockIdx.x * 8 + (threadIdx.x >> 5);
    int lane = threadIdx.x & 31;
    const float2* z2 = reinterpret_cast<const float2*>(z);
    float2 zz = z2[(size_t)i * 32 + lane];           // cols 2*lane, 2*lane+1
    float ps = fmaf(zz.x, a_src[2 * lane], zz.y * a_src[2 * lane + 1]);
    float pd = fmaf(zz.x, a_dst[2 * lane], zz.y * a_dst[2 * lane + 1]);
    float ss = wave_sum(ps);
    float sd = wave_sum(pd);
    if (lane == 0) {
        s_src[i] = ss;
        s_dst[i] = sd;
        denom[i] = 0.0f;
        segmax[i] = ENC_NEG_INF;
    }
    float2* acc2 = reinterpret_cast<float2*>(acc);
    acc2[(size_t)i * 32 + lane] = make_float2(0.0f, 0.0f);
}

// ---------- per-edge: score = lrelu(s_src[s]+s_dst[d]+t.a_t), segment max ----
// edge_t is a one-shot 205MB stream: load non-temporally so the hot
// (z/acc/denom/segmax/escore) working set stays resident in the 192MB L2.
__global__ void edgescore_kernel(const float* __restrict__ et,
                                 const float* __restrict__ a_t,
                                 const int* __restrict__ src,
                                 const int* __restrict__ dst,
                                 const float* __restrict__ s_src,
                                 const float* __restrict__ s_dst,
                                 float* __restrict__ escore,
                                 unsigned* __restrict__ segmax) {
    int e = blockIdx.x * 8 + (threadIdx.x >> 5);
    int lane = threadIdx.x & 31;
    const v2f* et2 = reinterpret_cast<const v2f*>(et);   // clang ext-vector: OK for NT builtin
    v2f t = __builtin_nontemporal_load(&et2[(size_t)e * 32 + lane]);
    float p = fmaf(t.x, a_t[2 * lane], t.y * a_t[2 * lane + 1]);
    float dot = wave_sum(p);
    if (lane == 0) {
        float sc = s_src[src[e]] + s_dst[dst[e]] + dot;
        sc = (sc > 0.0f) ? sc : 0.01f * sc;   // leaky_relu, slope 0.01
        escore[e] = sc;
        atomicMax(&segmax[dst[e]], enc_f(sc));
    }
}

// ---------- per-edge: ex=exp(score-m[d]); denom+=ex; acc[d] += ex*z[s] ------
__global__ void edgeaccum_kernel(const float* __restrict__ escore,
                                 const unsigned* __restrict__ segmax,
                                 const int* __restrict__ src,
                                 const int* __restrict__ dst,
                                 const float* __restrict__ z,
                                 float* __restrict__ acc,
                                 float* __restrict__ denom) {
    int e = blockIdx.x * 8 + (threadIdx.x >> 5);
    int lane = threadIdx.x & 31;
    int s = src[e];           // same address across lanes -> broadcast load
    int d = dst[e];
    float m = dec_f(segmax[d]);
    float ex = __expf(escore[e] - m);
    if (lane == 0) atomicAdd(&denom[d], ex);
    const float2* z2 = reinterpret_cast<const float2*>(z);
    float2 zz = z2[(size_t)s * 32 + lane];           // cols 2*lane, 2*lane+1
    atomicAdd(&acc[(size_t)d * HD + 2 * lane],     ex * zz.x);
    atomicAdd(&acc[(size_t)d * HD + 2 * lane + 1], ex * zz.y);
}

// ---------- per-element: out = acc/denom' + gate*v*zi ----------
__global__ void nodefinal_kernel(const float* __restrict__ accin,
                                 const float* __restrict__ denom,
                                 const float* __restrict__ gate,
                                 const float* __restrict__ v,
                                 const float* __restrict__ zi,
                                 float* __restrict__ out) {
    int idx = blockIdx.x * 256 + threadIdx.x;   // N*64 threads
    int i = idx >> 6, c = idx & 63;
    float den = denom[i];
    den = (den == 0.0f) ? 1.0f : den;
    out[idx] = accin[idx] / den + gate[c] * v[idx] * zi[idx];
}

extern "C" void kernel_launch(void* const* d_in, const int* in_sizes, int n_in,
                              void* d_out, int out_size, void* d_ws, size_t ws_size,
                              hipStream_t stream) {
    const float* attr    = (const float*)d_in[0];
    const float* pot0    = (const float*)d_in[1];
    const float* pot1    = (const float*)d_in[2];
    const float* edge_t0 = (const float*)d_in[3];
    const float* edge_t1 = (const float*)d_in[4];
    const float* Wpot    = (const float*)d_in[5];
    const float* bpot    = (const float*)d_in[6];
    const float* fcW0    = (const float*)d_in[7];
    const float* fcU0    = (const float*)d_in[8];
    const float* a0      = (const float*)d_in[9];
    const float* gate0   = (const float*)d_in[10];
    const float* fcW1    = (const float*)d_in[11];
    const float* fcU1    = (const float*)d_in[12];
    const float* a1      = (const float*)d_in[13];
    const float* gate1   = (const float*)d_in[14];
    const int*   src     = (const int*)d_in[15];
    const int*   dst     = (const int*)d_in[16];
    float* out = (float*)d_out;

    const size_t NN = (size_t)NNODES * HD;
    float* ws     = (float*)d_ws;
    float* z      = ws;            // N*64
    float* zi     = z + NN;        // N*64
    float* acc    = zi + NN;       // N*64 (also holds layer-1 output h, in place)
    float* vbuf   = acc + NN;      // N*64
    float* s_src  = vbuf + NN;     // N
    float* s_dst  = s_src + NNODES;
    float* denom  = s_dst + NNODES;
    unsigned* segmax = (unsigned*)(denom + NNODES);   // N
    float* escore = (float*)(segmax + NNODES);        // E

    const int gElem  = (NNODES * HD) / 256;   // 12500
    const int gGemm  = NNODES / 16;           // 3125
    const int gNode  = NNODES / 8;            // 6250
    const int gEdge  = NEDGES / 8;            // 100000

    // ---- layer 0 (h = attr) ----
    pot_kernel<<<gElem, 256, 0, stream>>>(pot0, Wpot, bpot, vbuf);
    gemm_kernel<<<gGemm, 256, 0, stream>>>(attr, fcW0, fcU0, z, zi);
    nodeprep_kernel<<<gNode, 256, 0, stream>>>(z, a0, a0 + HD, s_src, s_dst,
                                               acc, denom, segmax);
    edgescore_kernel<<<gEdge, 256, 0, stream>>>(edge_t0, a0 + 2 * HD, src, dst,
                                                s_src, s_dst, escore, segmax);
    edgeaccum_kernel<<<gEdge, 256, 0, stream>>>(escore, segmax, src, dst, z,
                                                acc, denom);
    nodefinal_kernel<<<gElem, 256, 0, stream>>>(acc, denom, gate0, vbuf, zi,
                                                acc /* h1 in place */);

    // ---- layer 1 (h = acc) ----
    pot_kernel<<<gElem, 256, 0, stream>>>(pot1, Wpot, bpot, vbuf);
    gemm_kernel<<<gGemm, 256, 0, stream>>>(acc, fcW1, fcU1, z, zi);
    nodeprep_kernel<<<gNode, 256, 0, stream>>>(z, a1, a1 + HD, s_src, s_dst,
                                               acc, denom, segmax);
    edgescore_kernel<<<gEdge, 256, 0, stream>>>(edge_t1, a1 + 2 * HD, src, dst,
                                                s_src, s_dst, escore, segmax);
    edgeaccum_kernel<<<gEdge, 256, 0, stream>>>(escore, segmax, src, dst, z,
                                                acc, denom);
    nodefinal_kernel<<<gElem, 256, 0, stream>>>(acc, denom, gate1, vbuf, zi,
                                                out);
}
